// MultiHeadCausalTensionLayer_24180665876915
// MI455X (gfx1250) — compile-verified, matrix operand order checked by reading, not run on hardware
//
#include <hip/hip_runtime.h>

#define B_  2
#define T_  2048
#define D_  1024
#define H_  16
#define HD_ 64
#define W_  32
#define M_  (B_*T_)   // 4096 rows
#define TT_ 64        // attention tokens per block

typedef __bf16 v8bf  __attribute__((ext_vector_type(8)));
typedef __bf16 v16bf __attribute__((ext_vector_type(16)));
typedef float  v8f   __attribute__((ext_vector_type(8)));

__device__ __forceinline__ float bf2f(unsigned short u){
  union { unsigned u; float f; } c; c.u = ((unsigned)u) << 16; return c.f;
}
__device__ __forceinline__ unsigned short f2bf(float f){
  union { float f; unsigned u; } c; c.f = f;
  unsigned r = c.u + 0x7fffu + ((c.u >> 16) & 1u);   // round-to-nearest-even
  return (unsigned short)(r >> 16);
}
__device__ __forceinline__ float bfe2f(__bf16 x){
  return bf2f(__builtin_bit_cast(unsigned short, x));
}
__device__ __forceinline__ v16bf cat8(v8bf a, v8bf b){
  return __builtin_shufflevector(a, b, 0,1,2,3,4,5,6,7,8,9,10,11,12,13,14,15);
}
// 16x32 bf16 WMMA A/B fragment from a row-major bf16 tile (stride elems).
// Lane%16 = matrix row, lane/16 selects K-half; two 8-elem chunks at k, k+16.
__device__ __forceinline__ v16bf frag_ld(const unsigned short* p, int row0, int stride){
  const int lane = threadIdx.x & 31;
  const unsigned short* r = p + (row0 + (lane & 15)) * stride + ((lane >> 4) << 3);
  v8bf lo = *(const v8bf*)r;
  v8bf hi = *(const v8bf*)(r + 16);
  return cat8(lo, hi);
}
__device__ __forceinline__ v8f wmma_bf16(v16bf a, v16bf b, v8f c){
  return __builtin_amdgcn_wmma_f32_16x16x32_bf16(false, a, false, b, (short)0, c, false, false);
}
// CDNA5 async copy: global -> LDS, 16B per lane, tracked by ASYNCcnt.
__device__ __forceinline__ void async_b128(void* lds_ptr, const void* gptr){
  unsigned lo = (unsigned)(unsigned long long)lds_ptr;   // low 32 bits give the LDS address
  asm volatile("global_load_async_to_lds_b128 %0, %1, off"
               :: "v"(lo), "v"((unsigned long long)gptr) : "memory");
}
__device__ __forceinline__ void wait_async0(){
  asm volatile("s_wait_asynccnt 0" ::: "memory");
}

// ---------------- Kernel 1: fp32 -> bf16 conversion ----------------
__global__ void cvt_f32_bf16(const float* __restrict__ in,
                             unsigned short* __restrict__ out, int n){
  for (int i = blockIdx.x*blockDim.x + threadIdx.x; i < n; i += gridDim.x*blockDim.x)
    out[i] = f2bf(in[i]);
}

// ---------------- Kernel 2: fused Q + KV GEMM (A @ W^T) ----------------
// C[4096, 3072]: n<1024 -> q (bf16), n>=1024 -> kv (fp32 for the VALU
// attention pass). 128x128 block tile, 8 waves, each wave 32x64
// (8 WMMA / K-step). Async-to-LDS double buffering, one barrier per K-step.
__global__ __launch_bounds__(256)
void gemm_qkv(const unsigned short* __restrict__ xb,
              const unsigned short* __restrict__ wqb,
              const unsigned short* __restrict__ wkvb,
              unsigned short* __restrict__ qb,
              float* __restrict__ kvf){
  __shared__ unsigned short As[2][128*32];
  __shared__ unsigned short Bs[2][128*32];
  const int tid  = threadIdx.x;
  const int wave = tid >> 5;
  const int lane = tid & 31;
  const int tileN = blockIdx.x * 128;
  const int tileM = blockIdx.y * 128;
  const bool isQ = (tileN < D_);
  const unsigned short* wbase = isQ ? wqb : wkvb;
  const int wrow0 = isQ ? tileN : (tileN - D_);
  const int m0 = (wave & 3) * 32;
  const int n0 = (wave >> 2) * 64;
  v8f acc[2][4] = {};

  const int lr = tid >> 2;             // rows 0..63 (x2 chunks below)
  const int lc = (tid & 3) << 3;       // k-chunk 0/8/16/24
  const unsigned short* ga = xb    + (tileM + lr)*D_ + lc;
  const unsigned short* gb = wbase + (wrow0 + lr)*D_ + lc;

  // prologue: fill buffer 0 (each thread: rows lr and lr+64, A and B)
  async_b128(&As[0][lr*32 + lc],      ga);
  async_b128(&As[0][(lr+64)*32 + lc], ga + 64*D_);
  async_b128(&Bs[0][lr*32 + lc],      gb);
  async_b128(&Bs[0][(lr+64)*32 + lc], gb + 64*D_);
  wait_async0();
  __syncthreads();

  for (int it = 0; it < D_/32; ++it){
    const int cur = it & 1, nxt = cur ^ 1;
    if (it + 1 < D_/32){
      const int kb = (it + 1) * 32;
      async_b128(&As[nxt][lr*32 + lc],      ga + kb);
      async_b128(&As[nxt][(lr+64)*32 + lc], ga + kb + 64*D_);
      async_b128(&Bs[nxt][lr*32 + lc],      gb + kb);
      async_b128(&Bs[nxt][(lr+64)*32 + lc], gb + kb + 64*D_);
    }
    v16bf a0 = frag_ld(As[cur], m0,      32);
    v16bf a1 = frag_ld(As[cur], m0 + 16, 32);
    v16bf b0 = frag_ld(Bs[cur], n0,      32);
    v16bf b1 = frag_ld(Bs[cur], n0 + 16, 32);
    v16bf b2 = frag_ld(Bs[cur], n0 + 32, 32);
    v16bf b3 = frag_ld(Bs[cur], n0 + 48, 32);
    acc[0][0] = wmma_bf16(a0, b0, acc[0][0]);
    acc[0][1] = wmma_bf16(a0, b1, acc[0][1]);
    acc[0][2] = wmma_bf16(a0, b2, acc[0][2]);
    acc[0][3] = wmma_bf16(a0, b3, acc[0][3]);
    acc[1][0] = wmma_bf16(a1, b0, acc[1][0]);
    acc[1][1] = wmma_bf16(a1, b1, acc[1][1]);
    acc[1][2] = wmma_bf16(a1, b2, acc[1][2]);
    acc[1][3] = wmma_bf16(a1, b3, acc[1][3]);
    wait_async0();        // own writes to buf[nxt] complete
    __syncthreads();      // everyone done reading buf[cur] & writing buf[nxt]
  }

  const int mr = 8*(lane >> 4);
  const int nn = lane & 15;
  #pragma unroll
  for (int mi = 0; mi < 2; ++mi){
    #pragma unroll
    for (int j = 0; j < 4; ++j){
      #pragma unroll
      for (int r = 0; r < 8; ++r){
        int gm = tileM + m0 + mi*16 + r + mr;
        int gn = tileN + n0 + j*16 + nn;
        if (isQ) qb [gm*D_     + gn]        = f2bf(acc[mi][j][r]);
        else     kvf[gm*(2*D_) + (gn - D_)] = acc[mi][j][r];
      }
    }
  }
}

// ---------------- Kernel 3: strictly-causal 32-token window attention ----
// One block = (b, h, 64-token tile). KV window [t0-32, t0+64) staged in LDS
// as fp32 (48 KB) via async-to-LDS -> inner loop is pure FMA, no converts.
__global__ __launch_bounds__(TT_)
void attn_win(const unsigned short* __restrict__ qb,
              const float* __restrict__ kvf,
              unsigned short* __restrict__ msgb){
  __shared__ float kvs[(TT_ + W_)*128];   // 96 positions x (64 k + 64 v) fp32
  const int TILES = T_/TT_;
  const int bx   = blockIdx.x;
  const int tile = bx % TILES;
  const int h    = (bx / TILES) % H_;
  const int b    = bx / (TILES*H_);
  const int t0   = tile * TT_;
  for (int c = threadIdx.x; c < (TT_ + W_)*32; c += TT_){  // b128 = 4 floats
    int pos = c >> 5;
    int col = (c & 31) << 2;
    int t   = t0 - W_ + pos;
    if (t >= 0){
      async_b128(&kvs[pos*128 + col],
                 kvf + ((size_t)(b*T_) + t)*(2*D_) + h*128 + col);
    } else {
      float4 z = {0.f, 0.f, 0.f, 0.f};
      *(float4*)&kvs[pos*128 + col] = z;
    }
  }
  wait_async0();
  __syncthreads();
  const int i = threadIdx.x;
  const int t = t0 + i;
  float qf[64];
  const unsigned short* qrow = qb + ((b*T_) + t)*D_ + h*HD_;
  #pragma unroll
  for (int c = 0; c < 8; ++c){
    v8bf q8 = *(const v8bf*)(qrow + c*8);
    #pragma unroll
    for (int e = 0; e < 8; ++e) qf[c*8+e] = bfe2f(q8[e]);
  }
  float msg[64];
  #pragma unroll
  for (int d = 0; d < 64; ++d) msg[d] = 0.f;
  for (int w = 0; w < W_; ++w){   // window t-32 .. t-1 -> LDS rows i..i+31
    const float* row = &kvs[(i + w)*128];
    float s = 0.f;
    #pragma unroll
    for (int c = 0; c < 16; ++c){
      float4 k4 = *(const float4*)(row + c*4);
      s += qf[c*4+0]*k4.x + qf[c*4+1]*k4.y + qf[c*4+2]*k4.z + qf[c*4+3]*k4.w;
    }
    float tau = 1.0f / (1.0f + __expf(-s * 0.125f));   // sigmoid(s/sqrt(64))
    #pragma unroll
    for (int c = 0; c < 16; ++c){
      float4 v4 = *(const float4*)(row + 64 + c*4);
      msg[c*4+0] += tau * v4.x;
      msg[c*4+1] += tau * v4.y;
      msg[c*4+2] += tau * v4.z;
      msg[c*4+3] += tau * v4.w;
    }
  }
  unsigned short* orow = msgb + ((b*T_) + t)*D_ + h*HD_;
  #pragma unroll
  for (int d = 0; d < 64; ++d) orow[d] = f2bf(msg[d]);
}

// ---------------- Kernel 4: y = x + msg @ wo^T, fused RMS-norm ----------
// 16 rows per block spanning full N=1024 so the row reduction stays on-chip.
__global__ __launch_bounds__(256)
void gemm_o_rms(const unsigned short* __restrict__ msgb,
                const unsigned short* __restrict__ wob,
                const float* __restrict__ x,
                const float* __restrict__ norm_scale,
                float* __restrict__ out){
  __shared__ unsigned short As[2][16*32];
  __shared__ float ssq[16];
  const int tid  = threadIdx.x;
  const int wave = tid >> 5;
  const int lane = tid & 31;
  const int m0   = blockIdx.x * 16;
  const int n0   = wave * 128;       // each wave owns 8 N-subtiles of 16
  v8f acc[8] = {};
  if (tid < 64){                     // prologue A tile (buffer 0)
    int row = tid >> 2, col = (tid & 3) << 3;
    async_b128(&As[0][row*32 + col], msgb + (m0 + row)*D_ + col);
  }
  wait_async0();
  __syncthreads();
  for (int it = 0; it < D_/32; ++it){
    const int cur = it & 1, nxt = cur ^ 1;
    const int kb = it * 32;
    if (it + 1 < D_/32 && tid < 64){
      int row = tid >> 2, col = (tid & 3) << 3;
      async_b128(&As[nxt][row*32 + col], msgb + (m0 + row)*D_ + kb + 32 + col);
    }
    v16bf a = frag_ld(As[cur], 0, 32);
    #pragma unroll
    for (int j = 0; j < 8; ++j){
      const unsigned short* wp = wob + (n0 + j*16 + (lane & 15))*D_ + kb + ((lane >> 4) << 3);
      v16bf bfr = cat8(*(const v8bf*)wp, *(const v8bf*)(wp + 16));
      acc[j] = wmma_bf16(a, bfr, acc[j]);
    }
    wait_async0();
    __syncthreads();
  }
  if (tid < 16) ssq[tid] = 0.f;
  __syncthreads();
  const int mr = 8*(lane >> 4);
  float part[8];
  #pragma unroll
  for (int r = 0; r < 8; ++r) part[r] = 0.f;
  #pragma unroll
  for (int j = 0; j < 8; ++j){
    int gn = n0 + j*16 + (lane & 15);
    #pragma unroll
    for (int r = 0; r < 8; ++r){
      int gm = m0 + r + mr;
      float y = acc[j][r] + x[gm*D_ + gn];   // residual into accumulator
      acc[j][r] = y;
      part[r] += y*y;
    }
  }
  #pragma unroll
  for (int r = 0; r < 8; ++r) atomicAdd(&ssq[r + mr], part[r]);  // ds_add_f32
  __syncthreads();
  float rinv[8];
  #pragma unroll
  for (int r = 0; r < 8; ++r) rinv[r] = rsqrtf(ssq[r + mr]*(1.0f/D_) + 1e-6f);
  #pragma unroll
  for (int j = 0; j < 8; ++j){
    int gn = n0 + j*16 + (lane & 15);
    float sc = norm_scale[gn];
    #pragma unroll
    for (int r = 0; r < 8; ++r){
      int gm = m0 + r + mr;
      out[gm*D_ + gn] = acc[j][r] * rinv[r] * sc;
    }
  }
}

extern "C" void kernel_launch(void* const* d_in, const int* in_sizes, int n_in,
                              void* d_out, int out_size, void* d_ws, size_t ws_size,
                              hipStream_t stream){
  const float* x   = (const float*)d_in[0];
  const float* wq  = (const float*)d_in[1];
  const float* wkv = (const float*)d_in[2];
  const float* wo  = (const float*)d_in[3];
  const float* ns  = (const float*)d_in[4];
  float* out = (float*)d_out;

  char* ws = (char*)d_ws;
  size_t off = 0;
  auto alloc = [&](size_t bytes)->void*{
    void* p = (void*)(ws + off);
    off += ((bytes + 255) & ~(size_t)255);
    return p;
  };
  unsigned short* xb   = (unsigned short*)alloc((size_t)M_*D_*2);    // 8 MB
  unsigned short* wqb  = (unsigned short*)alloc((size_t)D_*D_*2);    // 2 MB
  unsigned short* wkvb = (unsigned short*)alloc((size_t)2*D_*D_*2);  // 4 MB
  unsigned short* wob  = (unsigned short*)alloc((size_t)D_*D_*2);    // 2 MB
  unsigned short* qb   = (unsigned short*)alloc((size_t)M_*D_*2);    // 8 MB
  float*          kvf  = (float*)        alloc((size_t)M_*2*D_*4);   // 32 MB
  unsigned short* msgb = (unsigned short*)alloc((size_t)M_*D_*2);    // 8 MB
  if (off > ws_size) return;

  cvt_f32_bf16<<<1024, 256, 0, stream>>>(x,   xb,   M_*D_);
  cvt_f32_bf16<<<512,  256, 0, stream>>>(wq,  wqb,  D_*D_);
  cvt_f32_bf16<<<512,  256, 0, stream>>>(wkv, wkvb, 2*D_*D_);
  cvt_f32_bf16<<<512,  256, 0, stream>>>(wo,  wob,  D_*D_);
  gemm_qkv<<<dim3(3072/128, M_/128), 256, 0, stream>>>(xb, wqb, wkvb, qb, kvf);
  attn_win<<<B_*H_*(T_/TT_), TT_, 0, stream>>>(qb, kvf, msgb);
  gemm_o_rms<<<M_/16, 256, 0, stream>>>(msgb, wob, x, ns, out);
}
